// GCNEncoder_37177236914660
// MI455X (gfx1250) — compile-verified, matrix operand order checked by reading, not run on hardware
//
#include <hip/hip_runtime.h>
#include <hip/hip_bf16.h>

typedef __attribute__((ext_vector_type(16))) _Float16 v16h;
typedef __attribute__((ext_vector_type(8)))  float    v8f;

// ---------------------------------------------------------------------------
// K-offset within a 32-wide K-slab for the 16-bit WMMA A/B fragment layout
// (CDNA5 ISA §7.12.2): lanes 0-15 hold K=0..7 (v0-3) and K=16..23 (v4-7);
// lanes 16-31 hold K=8..15 and K=24..31.  p = element pair index (e>>1).
// ---------------------------------------------------------------------------
__device__ __forceinline__ int wmma_k_of(int lane, int e) {
    int p = e >> 1;
    return ((p >= 4) ? 16 : 0) + ((lane >> 4) << 3) + ((p & 3) << 1) + (e & 1);
}

// ---------------------------------------------------------------------------
// Edge index int64 -> int32 (halves index traffic in hot scatter loops)
// ---------------------------------------------------------------------------
__global__ void convert_edges(const long long* __restrict__ ei64,
                              int* __restrict__ src32, int* __restrict__ dst32, int E) {
    int i = blockIdx.x * blockDim.x + threadIdx.x;
    if (i < E) {
        src32[i] = (int)ei64[i];
        dst32[i] = (int)ei64[E + i];
    }
}

// ---------------------------------------------------------------------------
// In-degree (as float) and 1/max(deg,1)
// ---------------------------------------------------------------------------
__global__ void degree_kernel(const int* __restrict__ dst, float* __restrict__ deg, int E) {
    int e = blockIdx.x * blockDim.x + threadIdx.x;
    if (e < E) atomicAdd(&deg[dst[e]], 1.0f);
}

__global__ void invdeg_kernel(const float* __restrict__ deg, float* __restrict__ inv, int n) {
    int i = blockIdx.x * blockDim.x + threadIdx.x;
    if (i < n) inv[i] = 1.0f / fmaxf(deg[i], 1.0f);
}

// ---------------------------------------------------------------------------
// Scatter-add, layer 1: agg1[dst][f] += x[src][f], f<27, agg1 stride 32.
// One wave per edge: indices loaded once, lanes cover features.
// ---------------------------------------------------------------------------
__global__ void scatter1_kernel(const int* __restrict__ src, const int* __restrict__ dst,
                                const float* __restrict__ x, float* __restrict__ agg1, int E) {
    int wave = (blockIdx.x * blockDim.x + threadIdx.x) >> 5;
    int lane = threadIdx.x & 31;
    if (wave >= E) return;
    int s = src[wave], d = dst[wave];
    if (lane < 27) atomicAdd(&agg1[d * 32 + lane], x[s * 27 + lane]);
}

// ---------------------------------------------------------------------------
// Scatter-add, layer 2: agg2[dst][0:128] += h[src][0:128].
// One wave per edge; one b128 load per lane, then 4 float atomics.
// ---------------------------------------------------------------------------
__global__ void scatter2_kernel(const int* __restrict__ src, const int* __restrict__ dst,
                                const float* __restrict__ h, float* __restrict__ agg2, int E) {
    int wave = (blockIdx.x * blockDim.x + threadIdx.x) >> 5;
    int lane = threadIdx.x & 31;
    if (wave >= E) return;
    int s = src[wave], d = dst[wave];
    const float* hs = h + (long long)s * 128 + lane * 4;
    float*       ad = agg2 + (long long)d * 128 + lane * 4;
    float4 hv = *(const float4*)hs;                 // global_load_b128
    atomicAdd(&ad[0], hv.x);
    atomicAdd(&ad[1], hv.y);
    atomicAdd(&ad[2], hv.z);
    atomicAdd(&ad[3], hv.w);
}

// ---------------------------------------------------------------------------
// Pre-pack [W_l ; W_r] (K-concat, zero-padded) into per-lane WMMA B fragments:
// fragment (kt,nt) stored as 32 lanes x 16 halves, so the GEMM does a single
// aligned 32B v16h load per fragment.  Layer 1: Kcat=64, Nout=128.
// ---------------------------------------------------------------------------
__global__ void pack_w1(const float* __restrict__ W1l, const float* __restrict__ W1r,
                        _Float16* __restrict__ wpack) {
    int tid = blockIdx.x * blockDim.x + threadIdx.x;   // 16 frags * 512
    if (tid >= 16 * 512) return;
    int frag = tid >> 9;
    int lane = (tid >> 4) & 31;
    int e    = tid & 15;
    int kt = frag >> 3, nt = frag & 7;
    int k = kt * 32 + wmma_k_of(lane, e);
    int n = nt * 16 + (lane & 15);
    float v = 0.0f;
    if (k < 32) { if (k < 27) v = W1l[k * 128 + n]; }
    else        { int r = k - 32; if (r < 27) v = W1r[r * 128 + n]; }
    wpack[frag * 512 + lane * 16 + e] = (_Float16)v;
}

// Layer 2: Kcat=256 (128 | 128), Nout=64 -> 32 fragments, no padding needed.
__global__ void pack_w2(const float* __restrict__ W2l, const float* __restrict__ W2r,
                        _Float16* __restrict__ wpack) {
    int tid = blockIdx.x * blockDim.x + threadIdx.x;   // 32 frags * 512
    if (tid >= 32 * 512) return;
    int frag = tid >> 9;
    int lane = (tid >> 4) & 31;
    int e    = tid & 15;
    int kt = frag >> 2, nt = frag & 3;
    int k = kt * 32 + wmma_k_of(lane, e);
    int n = nt * 16 + (lane & 15);
    float v = (k < 128) ? W2l[k * 64 + n] : W2r[(k - 128) * 64 + n];
    wpack[frag * 512 + lane * 16 + e] = (_Float16)v;
}

// ---------------------------------------------------------------------------
// Layer 1 fused GEMM: h = relu([agg1*inv | x] @ [W1l;W1r] + b1),  [N,128]
// 8 waves/block, one 16-node tile per wave, A staged via LDS.
// Uniform full-tile fast path avoids per-element exec juggling in epilogue.
// ---------------------------------------------------------------------------
__global__ void sage_layer1(const float* __restrict__ x, const float* __restrict__ agg1,
                            const float* __restrict__ invdeg,
                            const _Float16* __restrict__ wpack, const float* __restrict__ b1,
                            float* __restrict__ h, int nNodes) {
    __shared__ float As[8][16][64];                     // 32 KB
    const int wave = threadIdx.x >> 5;
    const int lane = threadIdx.x & 31;
    const int base = (blockIdx.x * 8 + wave) * 16;
    const bool fullTile = (base + 16) <= nNodes;        // wave-uniform

    // Stage concat A rows: [mean-agg(27) pad5 | x(27) pad5]
    for (int idx = lane; idx < 16 * 64; idx += 32) {
        int row = idx >> 6, col = idx & 63;
        int node = base + row;
        float v = 0.0f;
        if (node < nNodes) {
            if (col < 27)                   v = agg1[node * 32 + col] * invdeg[node];
            else if (col >= 32 && col < 59) v = x[node * 27 + (col - 32)];
        }
        As[wave][row][col] = v;
    }
    __syncthreads();

    const int mrow = lane & 15;         // A row for this lane; also C column N
    const int hi8  = (lane >> 4) << 3;  // C/D row offset for hi lanes
    v16h a0, a1;
#pragma unroll
    for (int e = 0; e < 16; ++e) {
        int k = wmma_k_of(lane, e);
        a0[e] = (_Float16)As[wave][mrow][k];
        a1[e] = (_Float16)As[wave][mrow][32 + k];
    }

#pragma unroll
    for (int nt = 0; nt < 8; ++nt) {
        float bias = b1[nt * 16 + mrow];
        v8f c = {bias, bias, bias, bias, bias, bias, bias, bias};
        v16h bf0 = *(const v16h*)(wpack + (0 * 8 + nt) * 512 + lane * 16);
        v16h bf1 = *(const v16h*)(wpack + (1 * 8 + nt) * 512 + lane * 16);
        c = __builtin_amdgcn_wmma_f32_16x16x32_f16(false, a0, false, bf0, (short)0, c, false, false);
        c = __builtin_amdgcn_wmma_f32_16x16x32_f16(false, a1, false, bf1, (short)0, c, false, false);

        float* hp = h + (long long)(base + hi8) * 128 + nt * 16 + mrow;
        if (fullTile) {
#pragma unroll
            for (int r = 0; r < 8; ++r)
                hp[(long long)r * 128] = fmaxf(c[r], 0.0f);
        } else {
#pragma unroll
            for (int r = 0; r < 8; ++r)
                if (base + hi8 + r < nNodes)
                    hp[(long long)r * 128] = fmaxf(c[r], 0.0f);
        }
    }
}

// ---------------------------------------------------------------------------
// Layer 2 fused GEMM: out = [agg2*inv | h] @ [W2l;W2r] + b2,  [N,64]
// 2 waves/block (Kcat=256 -> 16 KB LDS per wave tile).
// ---------------------------------------------------------------------------
__global__ void sage_layer2(const float* __restrict__ h, const float* __restrict__ agg2,
                            const float* __restrict__ invdeg,
                            const _Float16* __restrict__ wpack, const float* __restrict__ b2,
                            float* __restrict__ out, int nNodes) {
    __shared__ float As[2][16][256];                    // 32 KB
    const int wave = threadIdx.x >> 5;
    const int lane = threadIdx.x & 31;
    const int base = (blockIdx.x * 2 + wave) * 16;
    const bool fullTile = (base + 16) <= nNodes;        // wave-uniform

    for (int idx = lane; idx < 16 * 256; idx += 32) {
        int row = idx >> 8, col = idx & 255;
        int node = base + row;
        float v = 0.0f;
        if (node < nNodes) {
            v = (col < 128) ? agg2[(long long)node * 128 + col] * invdeg[node]
                            : h[(long long)node * 128 + (col - 128)];
        }
        As[wave][row][col] = v;
    }
    __syncthreads();

    const int mrow = lane & 15;
    const int hi8  = (lane >> 4) << 3;
    v16h afrag[8];
#pragma unroll
    for (int kt = 0; kt < 8; ++kt)
#pragma unroll
        for (int e = 0; e < 16; ++e)
            afrag[kt][e] = (_Float16)As[wave][mrow][kt * 32 + wmma_k_of(lane, e)];

#pragma unroll
    for (int nt = 0; nt < 4; ++nt) {
        float bias = b2[nt * 16 + mrow];
        v8f c = {bias, bias, bias, bias, bias, bias, bias, bias};
#pragma unroll
        for (int kt = 0; kt < 8; ++kt) {
            v16h bf = *(const v16h*)(wpack + (kt * 4 + nt) * 512 + lane * 16);
            c = __builtin_amdgcn_wmma_f32_16x16x32_f16(false, afrag[kt], false, bf, (short)0, c, false, false);
        }
        float* op = out + (long long)(base + hi8) * 64 + nt * 16 + mrow;
        if (fullTile) {
#pragma unroll
            for (int r = 0; r < 8; ++r)
                op[(long long)r * 64] = c[r];
        } else {
#pragma unroll
            for (int r = 0; r < 8; ++r)
                if (base + hi8 + r < nNodes)
                    op[(long long)r * 64] = c[r];
        }
    }
}

// ---------------------------------------------------------------------------
// Host launcher
// ---------------------------------------------------------------------------
extern "C" void kernel_launch(void* const* d_in, const int* in_sizes, int n_in,
                              void* d_out, int out_size, void* d_ws, size_t ws_size,
                              hipStream_t stream) {
    const float*      x    = (const float*)d_in[0];
    const long long*  ei   = (const long long*)d_in[1];   // int64 [2,E]
    const float*      W1l  = (const float*)d_in[2];
    const float*      b1   = (const float*)d_in[3];
    const float*      W1r  = (const float*)d_in[4];
    const float*      W2l  = (const float*)d_in[5];
    const float*      b2   = (const float*)d_in[6];
    const float*      W2r  = (const float*)d_in[7];
    float*            out  = (float*)d_out;

    const int N = in_sizes[0] / 27;
    const int E = in_sizes[1] / 2;

    // Workspace carve-up (256B aligned regions)
    char*  ws  = (char*)d_ws;
    size_t off = 0;
    auto carve = [&](size_t bytes) {
        void* p = ws + off;
        off = (off + bytes + 255) & ~(size_t)255;
        return p;
    };
    int*       src32  = (int*)carve((size_t)E * 4);
    int*       dst32  = (int*)carve((size_t)E * 4);
    float*     deg    = (float*)carve((size_t)N * 4);
    float*     invd   = (float*)carve((size_t)N * 4);
    float*     agg1   = (float*)carve((size_t)N * 32 * 4);    // padded 27->32
    float*     hbuf   = (float*)carve((size_t)N * 128 * 4);
    float*     agg2   = (float*)carve((size_t)N * 128 * 4);
    _Float16*  wpack1 = (_Float16*)carve((size_t)16 * 512 * 2);
    _Float16*  wpack2 = (_Float16*)carve((size_t)32 * 512 * 2);

    // Zero accumulators (graph-capture-legal)
    hipMemsetAsync(deg,  0, (size_t)N * 4,        stream);
    hipMemsetAsync(agg1, 0, (size_t)N * 32 * 4,   stream);
    hipMemsetAsync(agg2, 0, (size_t)N * 128 * 4,  stream);

    const int T = 256;
    // Edge convert + degree
    convert_edges<<<(E + T - 1) / T, T, 0, stream>>>(ei, src32, dst32, E);
    degree_kernel<<<(E + T - 1) / T, T, 0, stream>>>(dst32, deg, E);
    invdeg_kernel<<<(N + T - 1) / T, T, 0, stream>>>(deg, invd, N);

    // Weight pre-pack into WMMA B-fragment layout
    pack_w1<<<(16 * 512 + T - 1) / T, T, 0, stream>>>(W1l, W1r, wpack1);
    pack_w2<<<(32 * 512 + T - 1) / T, T, 0, stream>>>(W2l, W2r, wpack2);

    const long long scatterBlocks = ((long long)E * 32 + T - 1) / T;
    const int tiles = (N + 15) / 16;

    // Layer 1: scatter (wave per edge) then fused WMMA GEMM + bias + relu
    scatter1_kernel<<<(unsigned)scatterBlocks, T, 0, stream>>>(src32, dst32, x, agg1, E);
    sage_layer1<<<(tiles + 7) / 8, 256, 0, stream>>>(x, agg1, invd, wpack1, b1, hbuf, N);

    // Layer 2: scatter h then fused WMMA GEMM + bias
    scatter2_kernel<<<(unsigned)scatterBlocks, T, 0, stream>>>(src32, dst32, hbuf, agg2, E);
    sage_layer2<<<(tiles + 1) / 2, 64, 0, stream>>>(hbuf, agg2, invd, wpack2, b2, out, N);
}
